// vv_11519102288412
// MI455X (gfx1250) — matrix operand order repaired and checked by reading, not compiled, not run on hardware
//
#include <hip/hip_runtime.h>
#include <hip/hip_bf16.h>
#include <math.h>

typedef __bf16 bf16_t;
typedef __attribute__((ext_vector_type(16))) __bf16 v16bf;
typedef __attribute__((ext_vector_type(8)))  __bf16 v8bf;
typedef __attribute__((ext_vector_type(8)))  float  v8f;
typedef __attribute__((ext_vector_type(4)))  unsigned int v4u;
typedef __attribute__((ext_vector_type(8)))  int v8i;
typedef __attribute__((ext_vector_type(4)))  int v4i;

#define QN    65536
#define HIDC  256
#define HEADS 16

#if defined(__has_builtin)
#  if __has_builtin(__builtin_amdgcn_tensor_load_to_lds)
#    define USE_TDM 1
#  endif
#endif
#ifndef USE_TDM
#  define USE_TDM 0
#endif

__device__ __forceinline__ float gelu_f(float x) {
  return 0.5f * x * (1.0f + erff(x * 0.70710678118654752f));
}
__device__ __forceinline__ bf16_t f2b(float x) { return (bf16_t)x; }
__device__ __forceinline__ float  b2f(bf16_t x) { return (float)x; }

// ---------------- conv 3x3: inp(3,64,64) -> feat(64,64,64) ----------------
__global__ __launch_bounds__(256) void k_conv_enc(const float* __restrict__ inp,
                                                  const float* __restrict__ w,
                                                  const float* __restrict__ b,
                                                  float* __restrict__ feat) {
  int gid = blockIdx.x * 256 + threadIdx.x;     // 64*4096 threads
  int px = gid & 4095, oc = gid >> 12;
  int y = px >> 6, x = px & 63;
  float acc = b[oc];
  for (int ic = 0; ic < 3; ++ic)
    for (int ky = 0; ky < 3; ++ky) {
      int yy = y + ky - 1; if (yy < 0 || yy > 63) continue;
      for (int kx = 0; kx < 3; ++kx) {
        int xx = x + kx - 1; if (xx < 0 || xx > 63) continue;
        acc += inp[ic * 4096 + yy * 64 + xx] * w[(oc * 3 + ic) * 9 + ky * 3 + kx];
      }
    }
  feat[oc * 4096 + px] = acc;
}

// ---------------- conv 3x3: feat(64,...) -> freq(256,...) -----------------
__global__ __launch_bounds__(256) void k_conv_freq(const float* __restrict__ feat,
                                                   const float* __restrict__ w,
                                                   const float* __restrict__ b,
                                                   float* __restrict__ freq) {
  __shared__ float s[576];                      // 9 taps x 64 in-channels
  int px = blockIdx.x;                          // 4096 blocks
  int y = px >> 6, x = px & 63;
  for (int i = threadIdx.x; i < 576; i += 256) {
    int t = i >> 6, ic = i & 63;
    int yy = y + t / 3 - 1, xx = x + t % 3 - 1;
    s[i] = (yy >= 0 && yy < 64 && xx >= 0 && xx < 64) ? feat[ic * 4096 + yy * 64 + xx] : 0.f;
  }
  __syncthreads();
  int oc = threadIdx.x;
  float acc = b[oc];
  for (int t = 0; t < 9; ++t)
    for (int ic = 0; ic < 64; ++ic)
      acc += s[t * 64 + ic] * w[oc * 576 + ic * 9 + t];
  freq[oc * 4096 + px] = acc;
}

// -------- gather: build grid0 (K=1056), grid1 (K=288), result_bil ---------
__global__ __launch_bounds__(256) void k_gather(const float* __restrict__ freq,
                                                const float* __restrict__ coord,
                                                const float* __restrict__ cell,
                                                const float* __restrict__ inp,
                                                bf16_t* __restrict__ grid0,
                                                bf16_t* __restrict__ grid1,
                                                float* __restrict__ bil) {
  int q = blockIdx.x * 256 + threadIdx.x;       // 65536 threads
  float cy = coord[q * 2 + 0], cx = coord[q * 2 + 1];
  float cely = cell[q * 2 + 0] * 64.f, celx = cell[q * 2 + 1] * 64.f;

  int iy[4], ix[4];
  float rely[4], relx[4], areas[4];
  float tot = 0.f;
  for (int s = 0; s < 4; ++s) {
    float dy = ((s < 2) ? -1.f : 1.f) * (1.f / 64.f) + 1e-6f;
    float dx = (((s & 1) == 0) ? -1.f : 1.f) * (1.f / 64.f) + 1e-6f;
    float sy = fminf(fmaxf(cy + dy, -1.f + 1e-6f), 1.f - 1e-6f);
    float sx = fminf(fmaxf(cx + dx, -1.f + 1e-6f), 1.f - 1e-6f);
    int yi = (int)rintf(((sy + 1.f) * 64.f - 1.f) * 0.5f);
    int xi = (int)rintf(((sx + 1.f) * 64.f - 1.f) * 0.5f);
    yi = min(max(yi, 0), 63); xi = min(max(xi, 0), 63);
    iy[s] = yi; ix[s] = xi;
    float qcy = -1.f + (2.f * yi + 1.f) / 64.f;
    float qcx = -1.f + (2.f * xi + 1.f) / 64.f;
    rely[s] = (cy - qcy) * 64.f;
    relx[s] = (cx - qcx) * 64.f;
    areas[s] = fabsf(rely[s] * relx[s]) + 1e-9f;
    tot += areas[s];
  }
  bf16_t* g0 = grid0 + (size_t)q * 1056;
  for (int s = 0; s < 4; ++s) { g0[2 * s] = f2b(rely[s]); g0[2 * s + 1] = f2b(relx[s]); }
  for (int s = 0; s < 4; ++s) {
    float wgt = areas[3 - s] / tot;
    const float* fr = freq + iy[s] * 64 + ix[s];
    for (int c = 0; c < 256; ++c) g0[8 + s * 256 + c] = f2b(fr[c * 4096] * wgt);
  }
  g0[1032] = f2b(cely); g0[1033] = f2b(celx);
  for (int c = 1034; c < 1056; ++c) g0[c] = f2b(0.f);

  // bilinear sample of freq at coord (zero outside, no border clamp)
  float fy = ((cy + 1.f) * 64.f - 1.f) * 0.5f;
  float fx = ((cx + 1.f) * 64.f - 1.f) * 0.5f;
  {
    float y0f = floorf(fy), x0f = floorf(fx);
    float wy = fy - y0f, wx = fx - x0f;
    int y0i = (int)y0f, x0i = (int)x0f;
    int ty[4] = { y0i, y0i, y0i + 1, y0i + 1 };
    int tx[4] = { x0i, x0i + 1, x0i, x0i + 1 };
    float tw[4] = { (1.f - wy) * (1.f - wx), (1.f - wy) * wx, wy * (1.f - wx), wy * wx };
    bf16_t* g1 = grid1 + (size_t)q * 288;
    g1[0] = f2b(cy); g1[1] = f2b(cx);
    for (int c = 0; c < 256; ++c) {
      float v = 0.f;
      for (int t = 0; t < 4; ++t)
        if (ty[t] >= 0 && ty[t] < 64 && tx[t] >= 0 && tx[t] < 64)
          v += tw[t] * freq[c * 4096 + ty[t] * 64 + tx[t]];
      g1[2 + c] = f2b(v);
    }
    g1[258] = f2b(cely); g1[259] = f2b(celx);
    for (int c = 260; c < 288; ++c) g1[c] = f2b(0.f);
  }
  // result_bil from inp, border=True
  {
    float by = fminf(fmaxf(fy, 0.f), 63.f);
    float bx = fminf(fmaxf(fx, 0.f), 63.f);
    float y0f = floorf(by), x0f = floorf(bx);
    float wy = by - y0f, wx = bx - x0f;
    int y0i = (int)y0f, x0i = (int)x0f;
    int ty[4] = { y0i, y0i, y0i + 1, y0i + 1 };
    int tx[4] = { x0i, x0i + 1, x0i, x0i + 1 };
    float tw[4] = { (1.f - wy) * (1.f - wx), (1.f - wy) * wx, wy * (1.f - wx), wy * wx };
    for (int c = 0; c < 3; ++c) {
      float v = 0.f;
      for (int t = 0; t < 4; ++t)
        if (ty[t] >= 0 && ty[t] < 64 && tx[t] >= 0 && tx[t] < 64)
          v += tw[t] * inp[c * 4096 + ty[t] * 64 + tx[t]];
      bil[q * 3 + c] = v;
    }
  }
}

// --------- weight pack: W(K x N, f32) -> WT(N x Kpad, bf16, zero-pad) -----
__global__ __launch_bounds__(256) void k_pack(const float* __restrict__ src,
                                              bf16_t* __restrict__ dst,
                                              int K, int N, int Kpad) {
  int i = blockIdx.x * 256 + threadIdx.x;
  if (i >= N * Kpad) return;
  int n = i / Kpad, k = i - n * Kpad;
  dst[i] = (k < K) ? f2b(src[(size_t)k * N + n]) : f2b(0.f);
}

// ------------------ bf16 WMMA GEMM: C = act(A*Bt + bias) + res ------------
// A:  (65536 x K) row-major bf16, lda = K (mult of 32)
// BT: (N x K) row-major bf16 (B transposed). Per block: 64-col panel of BT
//     staged into LDS (row pad +8 elems for bank stagger) via the Tensor
//     Data Mover (iterated 1-row tiles), then all 8 waves read B from LDS.
// block = 256 threads = 8 waves; wave computes 16 rows x 64 cols.
__global__ __launch_bounds__(256) void k_gemm(const bf16_t* __restrict__ A,
                                              const bf16_t* __restrict__ BT,
                                              const float* __restrict__ bias,
                                              const bf16_t* __restrict__ res,
                                              bf16_t* __restrict__ C,
                                              int N, int K, int act) {
  extern __shared__ __align__(16) char smem_raw[];
  bf16_t* sB = (bf16_t*)smem_raw;               // 64 rows x (K+8) bf16
  const int KP = K + 8;
  const int n0 = blockIdx.y * 64;

#if USE_TDM
  if (threadIdx.x < 32) {                       // wave 0 drives the TDM
    unsigned long long ga = (unsigned long long)(const void*)(BT + (size_t)n0 * K);
    unsigned td0 = (unsigned)K;
    v4u g0 = {};
    g0[0] = 1u;                                                  // count=1
    g0[1] = 0u;                                                  // lds_addr = 0
    g0[2] = (unsigned)(ga & 0xffffffffull);                      // global_addr lo
    g0[3] = (unsigned)((ga >> 32) & 0x01ffffffull) | (2u << 30); // addr hi | type=2
    v8i g1 = {};
    g1[0] = (int)((1u << 16) | (1u << 19));          // data_size=2B, iterate_en=1
    g1[1] = (int)((td0 & 0xffffu) << 16);            // tensor_dim0[15:0]
    g1[2] = (int)(((td0 >> 16) & 0xffffu) | (64u << 16)); // td0 hi | tensor_dim1=64
    g1[3] = (int)((td0 & 0xffffu) << 16);            // tile_dim0 = K
    g1[4] = 1;                                       // tile_dim1 = 1, tile_dim2 = 0
    g1[5] = (int)td0;                                // tensor_dim0_stride = K
    g1[6] = 0;
    g1[7] = 0;
    v4i g2 = {};
    g2[1] = (int)(unsigned)KP;                       // lds_addr_increment (elems)
    g2[2] = (int)td0;                                // global_addr_increment = K
    g2[3] = (int)(63u << 16);                        // iterate_count: 64 rows
    v4i g3 = {};
    v8i g4 = {};                                     // extra group (clang-23 form)
    __builtin_amdgcn_tensor_load_to_lds(g0, g1, g2, g3, g4, 0);
    __builtin_amdgcn_s_wait_tensorcnt(0);
  }
#else
  for (int i = threadIdx.x * 8; i < 64 * K; i += 256 * 8) {
    int row = i / K, col = i - (i / K) * K;
    *(v8bf*)(sB + row * KP + col) = *(const v8bf*)(BT + (size_t)(n0 + row) * K + col);
  }
#endif
  __syncthreads();

  int wave = threadIdx.x >> 5, lane = threadIdx.x & 31;
  int m0 = blockIdx.x * 128 + wave * 16;
  int r  = lane & 15, hh = lane >> 4;

  v8f acc[4] = {};
  const bf16_t* arow = A + (size_t)(m0 + r) * K;
  const bf16_t* sBl  = sB + (size_t)r * KP + 16 * hh;

  auto loadA = [&](int k0) -> v16bf {
    v8bf a0 = *(const v8bf*)(arow + k0 + 16 * hh);
    v8bf a1 = *(const v8bf*)(arow + k0 + 16 * hh + 8);
    return __builtin_shufflevector(a0, a1, 0,1,2,3,4,5,6,7,8,9,10,11,12,13,14,15);
  };

  v16bf a_cur = loadA(0);
  for (int k0 = 0; k0 < K; k0 += 32) {
    v16bf a_nxt = a_cur;                 // software pipeline: prefetch next A
    if (k0 + 32 < K) {
      a_nxt = loadA(k0 + 32);
      __builtin_prefetch(arow + k0 + 96, 0, 1);
    }
#pragma unroll
    for (int t = 0; t < 4; ++t) {
      v16bf bt = *(const v16bf*)(sBl + (size_t)t * 16 * KP + k0);
      acc[t] = __builtin_amdgcn_wmma_f32_16x16x32_bf16(
          false, a_cur, false, bt, (short)0, acc[t], false, false);
    }
    a_cur = a_nxt;
  }
#pragma unroll
  for (int t = 0; t < 4; ++t) {
    int col = n0 + t * 16 + r;
    float bv = bias ? bias[col] : 0.f;
#pragma unroll
    for (int v = 0; v < 8; ++v) {
      int row = m0 + v + 8 * hh;
      float val = acc[t][v] + bv;
      if (act) val = gelu_f(val);
      if (res) val += b2f(res[(size_t)row * N + col]);
      C[(size_t)row * N + col] = f2b(val);
    }
  }
}

// -------- per-row per-head layernorm of k,v; write transposed bf16 --------
__global__ __launch_bounds__(256) void k_ln(const bf16_t* __restrict__ qkv,
                                            const float* __restrict__ klw,
                                            const float* __restrict__ klb,
                                            const float* __restrict__ vlw,
                                            const float* __restrict__ vlb,
                                            bf16_t* __restrict__ klnT,
                                            bf16_t* __restrict__ vlnT) {
  int id = blockIdx.x * 256 + threadIdx.x;      // QN*HEADS threads
  int row = id >> 4, head = id & 15;
  const bf16_t* base = qkv + (size_t)row * 768 + head * 48;
  float kx[16], vx[16];
  float km = 0.f, vm = 0.f;
  for (int c = 0; c < 16; ++c) {
    kx[c] = b2f(base[16 + c]); vx[c] = b2f(base[32 + c]);
    km += kx[c]; vm += vx[c];
  }
  km *= (1.f / 16.f); vm *= (1.f / 16.f);
  float ks = 0.f, vs = 0.f;
  for (int c = 0; c < 16; ++c) {
    float dk = kx[c] - km, dv = vx[c] - vm;
    ks += dk * dk; vs += dv * dv;
  }
  float kstd = sqrtf(ks / 15.f), vstd = sqrtf(vs / 15.f);
  for (int c = 0; c < 16; ++c) {
    int ch = head * 16 + c;
    klnT[(size_t)ch * QN + row] = f2b(klw[ch] * ((kx[c] - km) / (kstd + 1e-5f)) + klb[ch]);
    vlnT[(size_t)ch * QN + row] = f2b(vlw[ch] * ((vx[c] - vm) / (vstd + 1e-5f)) + vlb[ch]);
  }
}

__global__ __launch_bounds__(256) void k_zero(float* __restrict__ p, int n) {
  int i = blockIdx.x * 256 + threadIdx.x;
  if (i < n) p[i] = 0.f;
}

// ------- kv[h,c,d] = (1/N) sum_rows kln[row,h,c]*vln[row,h,d] via WMMA ----
// klnT/vlnT: (256 channels x 65536 rows) bf16. 512 waves: head = wid/32.
__global__ __launch_bounds__(256) void k_kvreduce(const bf16_t* __restrict__ klnT,
                                                  const bf16_t* __restrict__ vlnT,
                                                  float* __restrict__ kv) {
  int wid  = blockIdx.x * 8 + (threadIdx.x >> 5);
  int head = wid >> 5, slot = wid & 31;
  int lane = threadIdx.x & 31;
  int r = lane & 15, hh = lane >> 4;
  const bf16_t* ka = klnT + (size_t)(head * 16 + r) * QN;
  const bf16_t* vb = vlnT + (size_t)(head * 16 + r) * QN;

  auto loadKA = [&](int k0) -> v16bf {
    v8bf a0 = *(const v8bf*)(ka + k0 + 16 * hh);
    v8bf a1 = *(const v8bf*)(ka + k0 + 16 * hh + 8);
    return __builtin_shufflevector(a0, a1, 0,1,2,3,4,5,6,7,8,9,10,11,12,13,14,15);
  };
  v8f acc = {};
  int base = slot * 2048;
  v16bf a_cur = loadKA(base);
  v16bf b_cur = *(const v16bf*)(vb + base + 16 * hh);
  for (int it = 0; it < 64; ++it) {
    v16bf a_nxt = a_cur, b_nxt = b_cur;
    if (it + 1 < 64) {
      int kn = base + (it + 1) * 32;
      a_nxt = loadKA(kn);
      b_nxt = *(const v16bf*)(vb + kn + 16 * hh);
    }
    acc = __builtin_amdgcn_wmma_f32_16x16x32_bf16(
        false, a_cur, false, b_cur, (short)0, acc, false, false);
    a_cur = a_nxt; b_cur = b_nxt;
  }
  const float scl = 1.f / (float)QN;
#pragma unroll
  for (int v = 0; v < 8; ++v)
    atomicAdd(&kv[head * 256 + (v + 8 * hh) * 16 + r], acc[v] * scl);
}

// -------------- attnv[row, h*16+d] = sum_c q[row,h,c]*kv[h,c,d] + x -------
__global__ __launch_bounds__(256) void k_apply(const bf16_t* __restrict__ qkv,
                                               const float* __restrict__ kv,
                                               const bf16_t* __restrict__ xin,
                                               bf16_t* __restrict__ attnv) {
  int id = blockIdx.x * 256 + threadIdx.x;
  int row = id >> 4, head = id & 15;
  const bf16_t* qb = qkv + (size_t)row * 768 + head * 48;
  float qv[16];
  for (int c = 0; c < 16; ++c) qv[c] = b2f(qb[c]);
  for (int d = 0; d < 16; ++d) {
    float s = 0.f;
    for (int c = 0; c < 16; ++c) s += qv[c] * kv[head * 256 + c * 16 + d];
    s += b2f(xin[(size_t)row * HIDC + head * 16 + d]);
    attnv[(size_t)row * HIDC + head * 16 + d] = f2b(s);
  }
}

__global__ __launch_bounds__(256) void k_add_bf(const bf16_t* __restrict__ a,
                                                const bf16_t* __restrict__ b,
                                                bf16_t* __restrict__ c) {
  size_t i = (size_t)blockIdx.x * 256 + threadIdx.x;
  c[i] = f2b(b2f(a[i]) + b2f(b[i]));
}

// --------- final head: out = t @ fc2_w2 + fc2_b2 + result_bil -------------
__global__ __launch_bounds__(256) void k_final(const bf16_t* __restrict__ t,
                                               const float* __restrict__ w2,
                                               const float* __restrict__ b2,
                                               const float* __restrict__ bil,
                                               float* __restrict__ out) {
  int q = blockIdx.x * 256 + threadIdx.x;
  const bf16_t* tr = t + (size_t)q * 256;
  float a0 = b2[0], a1 = b2[1], a2 = b2[2];
  for (int k = 0; k < 256; ++k) {
    float tv = b2f(tr[k]);
    a0 += tv * w2[k * 3 + 0];
    a1 += tv * w2[k * 3 + 1];
    a2 += tv * w2[k * 3 + 2];
  }
  out[q * 3 + 0] = a0 + bil[q * 3 + 0];
  out[q * 3 + 1] = a1 + bil[q * 3 + 1];
  out[q * 3 + 2] = a2 + bil[q * 3 + 2];
}

extern "C" void kernel_launch(void* const* d_in, const int* in_sizes, int n_in,
                              void* d_out, int out_size, void* d_ws, size_t ws_size,
                              hipStream_t stream) {
  (void)in_sizes; (void)n_in; (void)out_size; (void)ws_size;
  const float* inp    = (const float*)d_in[0];
  const float* coord  = (const float*)d_in[1];
  const float* cell   = (const float*)d_in[2];
  const float* enc_w  = (const float*)d_in[3];
  const float* enc_b  = (const float*)d_in[4];
  const float* freq_w = (const float*)d_in[5];
  const float* freq_b = (const float*)d_in[6];
  const float* out0_w1 = (const float*)d_in[7];
  const float* out0_b1 = (const float*)d_in[8];
  const float* out0_w2 = (const float*)d_in[9];
  const float* out0_b2 = (const float*)d_in[10];
  const float* out1_w1 = (const float*)d_in[11];
  const float* out1_b1 = (const float*)d_in[12];
  const float* out1_w2 = (const float*)d_in[13];
  const float* out1_b2 = (const float*)d_in[14];
  const float* fc1_w1 = (const float*)d_in[15];
  const float* fc1_b1 = (const float*)d_in[16];
  const float* fc1_w2 = (const float*)d_in[17];
  const float* fc1_b2 = (const float*)d_in[18];
  const float* fc2_w1 = (const float*)d_in[19];
  const float* fc2_b1 = (const float*)d_in[20];
  const float* fc2_w2 = (const float*)d_in[21];
  const float* fc2_b2 = (const float*)d_in[22];
  const float* aqkv_w[2]  = { (const float*)d_in[23], (const float*)d_in[31] };
  const float* aqkv_b[2]  = { (const float*)d_in[24], (const float*)d_in[32] };
  const float* akln_w[2]  = { (const float*)d_in[25], (const float*)d_in[33] };
  const float* akln_b[2]  = { (const float*)d_in[26], (const float*)d_in[34] };
  const float* avln_w[2]  = { (const float*)d_in[27], (const float*)d_in[35] };
  const float* avln_b[2]  = { (const float*)d_in[28], (const float*)d_in[36] };
  const float* aproj_w[2] = { (const float*)d_in[29], (const float*)d_in[37] };
  const float* aproj_b[2] = { (const float*)d_in[30], (const float*)d_in[38] };

  char* wsb = (char*)d_ws;
  size_t off = 0;
  auto carve = [&](size_t bytes) -> void* {
    void* p = wsb + off;
    off += (bytes + 255) & ~(size_t)255;
    return p;
  };
  float*  feat  = (float*)carve(64 * 4096 * 4);
  float*  freq  = (float*)carve(256 * 4096 * 4);
  bf16_t* grid0 = (bf16_t*)carve((size_t)QN * 1056 * 2);   // recycled later
  bf16_t* grid1 = (bf16_t*)carve((size_t)QN * 288 * 2);
  bf16_t* w01T  = (bf16_t*)carve(256 * 1056 * 2);
  bf16_t* w02T  = (bf16_t*)carve(256 * 256 * 2);
  bf16_t* w11T  = (bf16_t*)carve(256 * 288 * 2);
  bf16_t* w12T  = (bf16_t*)carve(256 * 256 * 2);
  bf16_t* qkvT[2], *projT[2];
  qkvT[0]  = (bf16_t*)carve(768 * 256 * 2);
  projT[0] = (bf16_t*)carve(256 * 256 * 2);
  qkvT[1]  = (bf16_t*)carve(768 * 256 * 2);
  projT[1] = (bf16_t*)carve(256 * 256 * 2);
  bf16_t* f11T  = (bf16_t*)carve(256 * 256 * 2);
  bf16_t* f12T  = (bf16_t*)carve(256 * 256 * 2);
  bf16_t* f21T  = (bf16_t*)carve(256 * 256 * 2);
  bf16_t* hbuf  = (bf16_t*)carve((size_t)QN * 256 * 2);
  bf16_t* x0in  = (bf16_t*)carve((size_t)QN * 256 * 2);
  bf16_t* x1in  = (bf16_t*)carve((size_t)QN * 256 * 2);
  bf16_t* vlnT  = (bf16_t*)carve((size_t)256 * QN * 2);
  bf16_t* attnv = (bf16_t*)carve((size_t)QN * 256 * 2);
  bf16_t* x0buf = (bf16_t*)carve((size_t)QN * 256 * 2);
  bf16_t* x1buf = (bf16_t*)carve((size_t)QN * 256 * 2);
  bf16_t* fbuf  = (bf16_t*)carve((size_t)QN * 256 * 2);
  bf16_t* tbuf  = (bf16_t*)carve((size_t)QN * 256 * 2);
  float*  kvbuf = (float*)carve(HEADS * 256 * 4);
  float*  bil   = (float*)carve((size_t)QN * 3 * 4);
  // recycle grid0 region (138.4 MB) after its one GEMM use:
  bf16_t* qkvbuf = grid0;                                   // 100.7 MB
  bf16_t* klnT   = (bf16_t*)((char*)grid0 + (size_t)QN * 768 * 2);  // +33.6 MB

  auto pack = [&](const float* src, bf16_t* dst, int K, int N, int Kpad) {
    int total = N * Kpad;
    k_pack<<<(total + 255) / 256, 256, 0, stream>>>(src, dst, K, N, Kpad);
  };
  auto gemm = [&](const bf16_t* A, int K, const bf16_t* BT, const float* bias,
                  const bf16_t* res, bf16_t* C, int N, int act) {
    dim3 g(QN / 128, N / 64);
    size_t sh = (size_t)64 * (K + 8) * sizeof(bf16_t);     // LDS B panel
    k_gemm<<<g, 256, sh, stream>>>(A, BT, bias, res, C, N, K, act);
  };

  // ---- weight packing ----
  pack(out0_w1, w01T, 1034, 256, 1056);
  pack(out0_w2, w02T, 256, 256, 256);
  pack(out1_w1, w11T, 260, 256, 288);
  pack(out1_w2, w12T, 256, 256, 256);
  pack(aqkv_w[0], qkvT[0], 256, 768, 256);
  pack(aproj_w[0], projT[0], 256, 256, 256);
  pack(aqkv_w[1], qkvT[1], 256, 768, 256);
  pack(aproj_w[1], projT[1], 256, 256, 256);
  pack(fc1_w1, f11T, 256, 256, 256);
  pack(fc1_w2, f12T, 256, 256, 256);
  pack(fc2_w1, f21T, 256, 256, 256);

  // ---- encoder convs ----
  k_conv_enc<<<1024, 256, 0, stream>>>(inp, enc_w, enc_b, feat);
  k_conv_freq<<<4096, 256, 0, stream>>>(feat, freq_w, freq_b, freq);

  // ---- sampling / feature assembly ----
  k_gather<<<QN / 256, 256, 0, stream>>>(freq, coord, cell, inp, grid0, grid1, bil);

  // ---- branch 0: mlp(grid0) ----
  gemm(grid0, 1056, w01T, out0_b1, nullptr, hbuf, 256, 1);
  gemm(hbuf, 256, w02T, out0_b2, nullptr, x0in, 256, 0);

  // ---- attn 0 ----  (grid0 region now dead -> reuse for qkvbuf/klnT)
  gemm(x0in, 256, qkvT[0], aqkv_b[0], nullptr, qkvbuf, 768, 0);
  k_ln<<<QN * HEADS / 256, 256, 0, stream>>>(qkvbuf, akln_w[0], akln_b[0],
                                             avln_w[0], avln_b[0], klnT, vlnT);
  k_zero<<<16, 256, 0, stream>>>(kvbuf, HEADS * 256);
  k_kvreduce<<<64, 256, 0, stream>>>(klnT, vlnT, kvbuf);
  k_apply<<<QN * HEADS / 256, 256, 0, stream>>>(qkvbuf, kvbuf, x0in, attnv);
  gemm(attnv, 256, projT[0], aproj_b[0], x0in, x0buf, 256, 1);

  // ---- branch 1: mlp(grid1) ----
  gemm(grid1, 288, w11T, out1_b1, nullptr, hbuf, 256, 1);
  gemm(hbuf, 256, w12T, out1_b2, nullptr, x1in, 256, 0);

  // ---- attn 1 ----
  gemm(x1in, 256, qkvT[1], aqkv_b[1], nullptr, qkvbuf, 768, 0);
  k_ln<<<QN * HEADS / 256, 256, 0, stream>>>(qkvbuf, akln_w[1], akln_b[1],
                                             avln_w[1], avln_b[1], klnT, vlnT);
  k_zero<<<16, 256, 0, stream>>>(kvbuf, HEADS * 256);
  k_kvreduce<<<64, 256, 0, stream>>>(klnT, vlnT, kvbuf);
  k_apply<<<QN * HEADS / 256, 256, 0, stream>>>(qkvbuf, kvbuf, x1in, attnv);
  gemm(attnv, 256, projT[1], aproj_b[1], x1in, x1buf, 256, 1);

  // ---- fusion head ----
  k_add_bf<<<QN * 256 / 256, 256, 0, stream>>>(x0buf, x1buf, fbuf);   // f = x0+x1
  gemm(fbuf, 256, f11T, fc1_b1, nullptr, hbuf, 256, 1);               // gelu(f@w1+b1)
  gemm(hbuf, 256, f12T, fc1_b2, fbuf, tbuf, 256, 1);                  // t = f + gelu(h@w2+b2)
  gemm(tbuf, 256, f21T, fc2_b1, nullptr, hbuf, 256, 1);               // gelu(t@fc2_w1+b1)
  k_final<<<QN / 256, 256, 0, stream>>>(hbuf, fc2_w2, fc2_b2, bil, (float*)d_out);
}
